// TransformerBlock_72971494359578
// MI455X (gfx1250) — matrix operand order repaired and checked by reading, not compiled, not run on hardware
//
#include <hip/hip_runtime.h>

#define DIM   1024
#define NHEAD 16
#define HD    64
#define MLPD  4096
#define SEQ   2048
#define BATCH 2
#define ROWS  (BATCH * SEQ)   // 4096

typedef __attribute__((ext_vector_type(16))) __bf16 bf16x16;
typedef __attribute__((ext_vector_type(8)))  float  floatx8;

union FragBF { bf16x16 v; unsigned short u[16]; uint4 x[2]; };
union U4U16 { uint4 u4; unsigned short s[8]; };

#define WMMA_BF16(A, B, C) \
  __builtin_amdgcn_wmma_f32_16x16x32_bf16(false, (A), false, (B), (short)0, (C), false, false)

// CDNA5 async memory->LDS copy (ASYNCcnt-tracked), ISA 15.18.3 op 98.
__device__ __forceinline__ void async_load_b128(unsigned lds_off, const void* gptr) {
  asm volatile("global_load_async_to_lds_b128 %0, %1, off"
               :: "v"(lds_off), "v"(gptr) : "memory");
}
__device__ __forceinline__ void wait_async0() {
  asm volatile("s_wait_asynccnt 0x0" ::: "memory");
}
__device__ __forceinline__ unsigned lds_addr(const void* p) {
  return (unsigned)(size_t)p;   // flat shared aperture: addr[31:0] = LDS offset
}

__device__ __forceinline__ unsigned short f2bf(float f) {
  unsigned int u = __float_as_uint(f);
  unsigned int r = u + 0x7FFFu + ((u >> 16) & 1u);   // round-to-nearest-even
  return (unsigned short)(r >> 16);
}

__device__ __forceinline__ float gelu_f(float v) {
  // branchless tanh-form GELU: v * (1 - 1/(exp(2u)+1)), u = 0.79788456(v + 0.044715 v^3)
  float e = __expf(1.5957691f * (v + 0.044715f * v * v * v));
  return v * (1.0f - 1.0f / (e + 1.0f));
}

// ---------------------------------------------------------------- weights cvt
__global__ void cvt_f32_bf16_kernel(const float* __restrict__ src,
                                    unsigned short* __restrict__ dst, int n) {
  int i = blockIdx.x * blockDim.x + threadIdx.x;
  int stride = gridDim.x * blockDim.x;
  for (; i < n; i += stride) dst[i] = f2bf(src[i]);
}

// ---------------------------------------------------------------- layernorm
__global__ __launch_bounds__(256)
void ln_bf16_kernel(const float* __restrict__ x, const float* __restrict__ g,
                    const float* __restrict__ b, unsigned short* __restrict__ out) {
  __shared__ float s1[256];
  __shared__ float s2[256];
  int row = blockIdx.x;
  int t = threadIdx.x;
  const float* xr = x + (size_t)row * DIM;
  float v0 = xr[t], v1 = xr[t + 256], v2 = xr[t + 512], v3 = xr[t + 768];
  s1[t] = v0 + v1 + v2 + v3;
  s2[t] = v0 * v0 + v1 * v1 + v2 * v2 + v3 * v3;
  __syncthreads();
  for (int o = 128; o > 0; o >>= 1) {
    if (t < o) { s1[t] += s1[t + o]; s2[t] += s2[t + o]; }
    __syncthreads();
  }
  float mu = s1[0] * (1.0f / DIM);
  float var = s2[0] * (1.0f / DIM) - mu * mu;
  float rinv = rsqrtf(var + 1e-5f);
  unsigned short* orow = out + (size_t)row * DIM;
  orow[t]       = f2bf((v0 - mu) * rinv * g[t]       + b[t]);
  orow[t + 256] = f2bf((v1 - mu) * rinv * g[t + 256] + b[t + 256]);
  orow[t + 512] = f2bf((v2 - mu) * rinv * g[t + 512] + b[t + 512]);
  orow[t + 768] = f2bf((v3 - mu) * rinv * g[t + 768] + b[t + 768]);
}

// ---------------------------------------------------------------- GEMM
// C[M,N] = act(A_bf16[M,K] @ W_bf16[K,N] + bias) (+ residual), out fp32 and/or bf16.
// 256 threads (8 waves), block tile 128(M) x 64(N); wave (mw 0..3, nw 0..1) owns a
// 32x32 patch = 4 WMMA tiles per k-step of 32. DOUBLE-BUFFERED LDS: next A tile
// streamed with global_load_async_to_lds_b128 (ASYNCcnt) while WMMAs run on the
// current tile; W tile fetched to regs early and transpose-scattered after compute.
__global__ __launch_bounds__(256)
void gemm_bf16_kernel(const unsigned short* __restrict__ A,
                      const unsigned short* __restrict__ W,
                      const float* __restrict__ bias,
                      const float* __restrict__ residual,
                      float* __restrict__ outF,
                      unsigned short* __restrict__ outB,
                      int M, int N, int K, int act) {
  __shared__ __align__(16) unsigned short As[2][128 * 32];
  __shared__ __align__(16) unsigned short BsT[2][64 * 32];   // [n][k]

  int tid  = threadIdx.x;
  int lane = tid & 31;
  int wave = tid >> 5;
  int mw   = wave >> 1;            // 0..3
  int nw   = wave & 1;             // 0..1
  int mBlock = blockIdx.y * 128;
  int nBlock = blockIdx.x * 64;

  int l15   = lane & 15;
  int hi    = (lane >= 16);
  int koffA = hi ? 16 : 0;         // 16-bit A layout: contiguous k = 16h + s
  int koffB = hi ? 16 : 0;         // 16-bit B layout

  int aRow = tid >> 1;             // 0..127
  int aCol = (tid & 1) * 16;       // 0,16
  int kRow = tid & 31;             // 0..31
  int nGrp = tid >> 5;             // 0..7 -> n = nGrp*8..+7

  floatx8 acc00 = {}, acc01 = {}, acc10 = {}, acc11 = {};

  // ---- prologue: stage k-step 0 into buffer 0 ----
  {
    const unsigned short* ag = &A[(size_t)(mBlock + aRow) * K + aCol];
    unsigned lds = lds_addr(&As[0][aRow * 32 + aCol]);
    async_load_b128(lds, ag);
    async_load_b128(lds + 16, ag + 8);
    U4U16 wv;
    wv.u4 = *(const uint4*)(&W[(size_t)kRow * N + nBlock + nGrp * 8]);
#pragma unroll
    for (int j = 0; j < 8; ++j) BsT[0][(nGrp * 8 + j) * 32 + kRow] = wv.s[j];
    wait_async0();
    __syncthreads();
  }

  int cur = 0;
  for (int k0 = 0; k0 < K; k0 += 32) {
    int nxt = cur ^ 1;
    bool more = (k0 + 32) < K;
    U4U16 wv;
    if (more) {
      const unsigned short* ag = &A[(size_t)(mBlock + aRow) * K + (k0 + 32) + aCol];
      unsigned lds = lds_addr(&As[nxt][aRow * 32 + aCol]);
      async_load_b128(lds, ag);
      async_load_b128(lds + 16, ag + 8);
      wv.u4 = *(const uint4*)(&W[(size_t)(k0 + 32 + kRow) * N + nBlock + nGrp * 8]);
      if (k0 + 64 < K)
        __builtin_prefetch(&W[(size_t)(k0 + 64 + kRow) * N + nBlock + nGrp * 8], 0, 1);
    }

    // ---- fragments + WMMA from buffer `cur` ----
    FragBF a0, a1, b0, b1;
    int r0 = mw * 32 + l15;
    int r1 = r0 + 16;
    a0.x[0] = *(const uint4*)(&As[cur][r0 * 32 + koffA]);
    a0.x[1] = *(const uint4*)(&As[cur][r0 * 32 + koffA + 8]);
    a1.x[0] = *(const uint4*)(&As[cur][r1 * 32 + koffA]);
    a1.x[1] = *(const uint4*)(&As[cur][r1 * 32 + koffA + 8]);
    int c0 = nw * 32 + l15;
    int c1 = c0 + 16;
    b0.x[0] = *(const uint4*)(&BsT[cur][c0 * 32 + koffB]);
    b0.x[1] = *(const uint4*)(&BsT[cur][c0 * 32 + koffB + 8]);
    b1.x[0] = *(const uint4*)(&BsT[cur][c1 * 32 + koffB]);
    b1.x[1] = *(const uint4*)(&BsT[cur][c1 * 32 + koffB + 8]);

    acc00 = WMMA_BF16(a0.v, b0.v, acc00);
    acc01 = WMMA_BF16(a0.v, b1.v, acc01);
    acc10 = WMMA_BF16(a1.v, b0.v, acc10);
    acc11 = WMMA_BF16(a1.v, b1.v, acc11);

    if (more) {
#pragma unroll
      for (int j = 0; j < 8; ++j) BsT[nxt][(nGrp * 8 + j) * 32 + kRow] = wv.s[j];
      wait_async0();
      __syncthreads();
    }
    cur = nxt;
  }

  int colA = nBlock + nw * 32 + l15;
  int colB = colA + 16;
  float bvA = bias[colA];
  float bvB = bias[colB];
  int rowHalf = hi ? 8 : 0;

#pragma unroll
  for (int r = 0; r < 8; ++r) {
    int row0 = mBlock + mw * 32 + r + rowHalf;
    int row1 = row0 + 16;
    float v00 = acc00[r] + bvA;
    float v01 = acc01[r] + bvB;
    float v10 = acc10[r] + bvA;
    float v11 = acc11[r] + bvB;
    if (act == 1) { v00 = gelu_f(v00); v01 = gelu_f(v01); v10 = gelu_f(v10); v11 = gelu_f(v11); }
    if (residual) {
      v00 += residual[(size_t)row0 * N + colA];
      v01 += residual[(size_t)row0 * N + colB];
      v10 += residual[(size_t)row1 * N + colA];
      v11 += residual[(size_t)row1 * N + colB];
    }
    if (outF) {
      outF[(size_t)row0 * N + colA] = v00;
      outF[(size_t)row0 * N + colB] = v01;
      outF[(size_t)row1 * N + colA] = v10;
      outF[(size_t)row1 * N + colB] = v11;
    }
    if (outB) {
      outB[(size_t)row0 * N + colA] = f2bf(v00);
      outB[(size_t)row0 * N + colB] = f2bf(v01);
      outB[(size_t)row1 * N + colA] = f2bf(v10);
      outB[(size_t)row1 * N + colB] = f2bf(v11);
    }
  }
}

// ---------------------------------------------------------------- attention
// qkv layout: [B*L, 3*DIM] bf16, cols: q = h*64+d, k = DIM+h*64+d, v = 2*DIM+h*64+d.
// Block = 128 queries of one (batch, head); 8 waves, one 16-query tile each.
// K chunk staged via async-to-LDS; V chunk transpose-scattered; all fragments
// are pairs of contiguous b128 LDS loads.
__global__ __launch_bounds__(256)
void attn_kernel(const unsigned short* __restrict__ qkv,
                 unsigned short* __restrict__ out) {
  __shared__ __align__(16) unsigned short Ks[32 * 64];    // [key][d]
  __shared__ __align__(16) unsigned short VsT[64 * 32];   // [d][key]
  __shared__ __align__(16) unsigned short Ps[8][16 * 32]; // per-wave P tile

  int tid   = threadIdx.x;
  int lane  = tid & 31;
  int wave  = tid >> 5;
  int head  = blockIdx.y;
  int batch = blockIdx.z;
  int qbase = blockIdx.x * 128 + wave * 16;
  int l15   = lane & 15;
  int hi    = (lane >= 16);
  const int LD = 3 * DIM;
  size_t rowBase = (size_t)batch * SEQ;

  // staging indices
  int sRow = tid >> 3;        // 0..31 (key within chunk)
  int sCol = (tid & 7) * 8;   // d

  // Q fragments (A layout): two contiguous K-runs per lane
  FragBF q0, q1;
  {
    const unsigned short* qp =
        qkv + (rowBase + qbase + l15) * LD + head * HD;
    int koff = hi ? 16 : 0;
    q0.x[0] = *(const uint4*)(qp + koff);
    q0.x[1] = *(const uint4*)(qp + koff + 8);
    q1.x[0] = *(const uint4*)(qp + 32 + koff);
    q1.x[1] = *(const uint4*)(qp + 32 + koff + 8);
  }

  floatx8 o0 = {}, o1 = {}, o2 = {}, o3 = {};
  float m[8], lsum[8];
#pragma unroll
  for (int r = 0; r < 8; ++r) { m[r] = -3.0e38f; lsum[r] = 0.0f; }

  const float scale = 0.125f;                 // 1/sqrt(64)
  int kend = blockIdx.x * 128 + 128;          // block-uniform causal frontier

  for (int kb = 0; kb < kend; kb += 32) {
    // ---- stage K via async-to-LDS, V transposed via regs ----
    const unsigned short* kg =
        qkv + (rowBase + kb + sRow) * LD + DIM + head * HD + sCol;
    async_load_b128(lds_addr(&Ks[sRow * 64 + sCol]), kg);
    U4U16 vv;
    vv.u4 = *(const uint4*)(qkv + (rowBase + kb + sRow) * LD + 2 * DIM + head * HD + sCol);
#pragma unroll
    for (int j = 0; j < 8; ++j) VsT[(sCol + j) * 32 + sRow] = vv.s[j];
    if (kb + 32 < kend) {
      __builtin_prefetch(qkv + (rowBase + kb + 32 + sRow) * LD + DIM + head * HD + sCol, 0, 1);
      __builtin_prefetch(qkv + (rowBase + kb + 32 + sRow) * LD + 2 * DIM + head * HD + sCol, 0, 1);
    }
    wait_async0();
    __syncthreads();

    // ---- S = Q @ K^T for 32 keys (two 16x16 C tiles) ----
    int doff = hi ? 16 : 0;
    FragBF b00, b01, b10, b11;   // [dstep][keysub]
    {
      int kr0 = (0 * 16 + l15) * 64;   // key sub 0 row base
      int kr1 = (1 * 16 + l15) * 64;   // key sub 1 row base
      b00.x[0] = *(const uint4*)(&Ks[kr0 + doff]);
      b00.x[1] = *(const uint4*)(&Ks[kr0 + doff + 8]);
      b01.x[0] = *(const uint4*)(&Ks[kr1 + doff]);
      b01.x[1] = *(const uint4*)(&Ks[kr1 + doff + 8]);
      b10.x[0] = *(const uint4*)(&Ks[kr0 + 32 + doff]);
      b10.x[1] = *(const uint4*)(&Ks[kr0 + 32 + doff + 8]);
      b11.x[0] = *(const uint4*)(&Ks[kr1 + 32 + doff]);
      b11.x[1] = *(const uint4*)(&Ks[kr1 + 32 + doff + 8]);
    }
    floatx8 s0 = {}, s1 = {};
    s0 = WMMA_BF16(q0.v, b00.v, s0);
    s0 = WMMA_BF16(q1.v, b10.v, s0);
    s1 = WMMA_BF16(q0.v, b01.v, s1);
    s1 = WMMA_BF16(q1.v, b11.v, s1);

    // ---- online softmax update ----
#pragma unroll
    for (int r = 0; r < 8; ++r) {
      int q = qbase + r + (hi ? 8 : 0);
      float sv0 = s0[r] * scale;
      float sv1 = s1[r] * scale;
      sv0 = (kb + l15 > q)      ? -3.0e38f : sv0;
      sv1 = (kb + 16 + l15 > q) ? -3.0e38f : sv1;
      float t = fmaxf(sv0, sv1);
      for (int mk = 1; mk < 16; mk <<= 1) t = fmaxf(t, __shfl_xor(t, mk, 32));
      float mnew = fmaxf(m[r], t);
      float corr = __expf(m[r] - mnew);
      float p0 = __expf(sv0 - mnew);
      float p1 = __expf(sv1 - mnew);
      float rs = p0 + p1;
      for (int mk = 1; mk < 16; mk <<= 1) rs += __shfl_xor(rs, mk, 32);
      lsum[r] = lsum[r] * corr + rs;
      m[r] = mnew;
      o0[r] *= corr; o1[r] *= corr; o2[r] *= corr; o3[r] *= corr;
      int prow = r + (hi ? 8 : 0);
      Ps[wave][prow * 32 + l15]      = f2bf(p0);
      Ps[wave][prow * 32 + 16 + l15] = f2bf(p1);
    }

    // ---- O += P @ V ----
    FragBF pf;
    {
      int koff = hi ? 16 : 0;
      pf.x[0] = *(const uint4*)(&Ps[wave][l15 * 32 + koff]);
      pf.x[1] = *(const uint4*)(&Ps[wave][l15 * 32 + koff + 8]);
    }
    int koff2 = hi ? 16 : 0;
    FragBF v0f, v1f, v2f, v3f;
    v0f.x[0] = *(const uint4*)(&VsT[(0 * 16 + l15) * 32 + koff2]);
    v0f.x[1] = *(const uint4*)(&VsT[(0 * 16 + l15) * 32 + koff2 + 8]);
    v1f.x[0] = *(const uint4*)(&VsT[(1 * 16 + l15) * 32 + koff2]);
    v1f.x[1] = *(const uint4*)(&VsT[(1 * 16 + l15) * 32 + koff2 + 8]);
    v2f.x[0] = *(const uint4*)(&VsT[(2 * 16 + l15) * 32 + koff2]);
    v2f.x[1] = *(const uint4*)(&VsT[(2 * 16 + l15) * 32 + koff2 + 8]);
    v3f.x[0] = *(const uint4*)(&VsT[(3 * 16 + l15) * 32 + koff2]);
    v3f.x[1] = *(const uint4*)(&VsT[(3 * 16 + l15) * 32 + koff2 + 8]);

    o0 = WMMA_BF16(pf.v, v0f.v, o0);
    o1 = WMMA_BF16(pf.v, v1f.v, o1);
    o2 = WMMA_BF16(pf.v, v2f.v, o2);
    o3 = WMMA_BF16(pf.v, v3f.v, o3);
    __syncthreads();
  }

  // ---- finalize ----
  int col0 = head * HD + l15;
#pragma unroll
  for (int r = 0; r < 8; ++r) {
    int row = qbase + r + (hi ? 8 : 0);
    float inv = 1.0f / lsum[r];
    unsigned short* op = out + (rowBase + row) * DIM;
    op[col0]      = f2bf(o0[r] * inv);
    op[col0 + 16] = f2bf(o1[r] * inv);
    op[col0 + 32] = f2bf(o2[r] * inv);
    op[col0 + 48] = f2bf(o3[r] * inv);
  }
}

// ---------------------------------------------------------------- launcher
extern "C" void kernel_launch(void* const* d_in, const int* in_sizes, int n_in,
                              void* d_out, int out_size, void* d_ws, size_t ws_size,
                              hipStream_t stream) {
  (void)in_sizes; (void)n_in; (void)out_size; (void)ws_size;

  const float* x      = (const float*)d_in[0];
  const float* w_qkv  = (const float*)d_in[2];
  const float* b_qkv  = (const float*)d_in[3];
  const float* w_proj = (const float*)d_in[4];
  const float* b_proj = (const float*)d_in[5];
  const float* w_fc1  = (const float*)d_in[6];
  const float* b_fc1  = (const float*)d_in[7];
  const float* w_fc2  = (const float*)d_in[8];
  const float* b_fc2  = (const float*)d_in[9];
  const float* g1     = (const float*)d_in[10];
  const float* beta1  = (const float*)d_in[11];
  const float* g2     = (const float*)d_in[12];
  const float* beta2  = (const float*)d_in[13];
  float* out = (float*)d_out;

  char* ws = (char*)d_ws;
  size_t off = 0;
  auto carve = [&](size_t bytes) -> void* {
    void* p = ws + off;
    off = (off + bytes + 255) & ~(size_t)255;
    return p;
  };

  unsigned short* wqkv_bf = (unsigned short*)carve((size_t)DIM * 3 * DIM * 2);
  unsigned short* wprj_bf = (unsigned short*)carve((size_t)DIM * DIM * 2);
  unsigned short* wfc1_bf = (unsigned short*)carve((size_t)DIM * MLPD * 2);
  unsigned short* wfc2_bf = (unsigned short*)carve((size_t)MLPD * DIM * 2);
  unsigned short* h1      = (unsigned short*)carve((size_t)ROWS * DIM * 2);
  unsigned short* qkvb    = (unsigned short*)carve((size_t)ROWS * 3 * DIM * 2);
  unsigned short* attnb   = (unsigned short*)carve((size_t)ROWS * DIM * 2);
  float*          x2      = (float*)carve((size_t)ROWS * DIM * 4);
  unsigned short* h2      = (unsigned short*)carve((size_t)ROWS * DIM * 2);
  unsigned short* mlpb    = (unsigned short*)carve((size_t)ROWS * MLPD * 2);

  // 1) weights -> bf16
  cvt_f32_bf16_kernel<<<2048, 256, 0, stream>>>(w_qkv,  wqkv_bf, DIM * 3 * DIM);
  cvt_f32_bf16_kernel<<<2048, 256, 0, stream>>>(w_proj, wprj_bf, DIM * DIM);
  cvt_f32_bf16_kernel<<<2048, 256, 0, stream>>>(w_fc1,  wfc1_bf, DIM * MLPD);
  cvt_f32_bf16_kernel<<<2048, 256, 0, stream>>>(w_fc2,  wfc2_bf, MLPD * DIM);

  // 2) LN1
  ln_bf16_kernel<<<ROWS, 256, 0, stream>>>(x, g1, beta1, h1);

  // 3) QKV GEMM -> bf16
  gemm_bf16_kernel<<<dim3(3 * DIM / 64, ROWS / 128), 256, 0, stream>>>(
      h1, wqkv_bf, b_qkv, nullptr, nullptr, qkvb, ROWS, 3 * DIM, DIM, 0);

  // 4) causal attention
  attn_kernel<<<dim3(SEQ / 128, NHEAD, BATCH), 256, 0, stream>>>(qkvb, attnb);

  // 5) proj GEMM + residual(x) -> fp32 x2
  gemm_bf16_kernel<<<dim3(DIM / 64, ROWS / 128), 256, 0, stream>>>(
      attnb, wprj_bf, b_proj, x, x2, nullptr, ROWS, DIM, DIM, 0);

  // 6) LN2
  ln_bf16_kernel<<<ROWS, 256, 0, stream>>>(x2, g2, beta2, h2);

  // 7) FC1 GEMM + GELU -> bf16
  gemm_bf16_kernel<<<dim3(MLPD / 64, ROWS / 128), 256, 0, stream>>>(
      h2, wfc1_bf, b_fc1, nullptr, nullptr, mlpb, ROWS, MLPD, DIM, 1);

  // 8) FC2 GEMM + residual(x2) -> out
  gemm_bf16_kernel<<<dim3(DIM / 64, ROWS / 128), 256, 0, stream>>>(
      mlpb, wfc2_bf, b_fc2, x2, out, nullptr, ROWS, DIM, MLPD, 0);
}
